// EdgeFocusedAtt_27676769256046
// MI455X (gfx1250) — matrix operand order repaired and checked by reading, not compiled, not run on hardware
//
#include <hip/hip_runtime.h>
#include <hip/hip_bf16.h>

// ---------- types for WMMA fragments ----------
typedef __attribute__((ext_vector_type(16))) __bf16       v16bf;
typedef __attribute__((ext_vector_type(8)))  float        v8f;
typedef __attribute__((ext_vector_type(4)))  unsigned int v4u;
typedef __attribute__((ext_vector_type(8)))  unsigned int v8u;

union FragB16 { v8u u; v16bf b; };

// dims
#define BATCH 16
#define CH    128
#define HW    9216          // 96*96
#define MTOT  2048          // BATCH*CH

// LDS tile geometry: 128 rows x 32 K-elements (64B payload), 80B pitch to kill
// bank conflicts on 16-lane ds_load_b128 (20*n mod 64 is a permutation).
#define ROWS   40           // row pitch in ushorts (80 bytes)
#define TILE_S (128 * ROWS) // one tile in ushorts (10240 B)
#define BUF_S  (2 * TILE_S) // A tile + B tile per buffer

__device__ __forceinline__ unsigned short f32_to_bf16(float f) {
  unsigned int u = __float_as_uint(f);
  u += 0x7FFFu + ((u >> 16) & 1u);        // round-to-nearest-even
  return (unsigned short)(u >> 16);
}

// ---------- CDNA5 async copy: global -> LDS, 16B per lane, tracked by ASYNCcnt ----
__device__ __forceinline__ void async_b128(unsigned lds_off_bytes, const void* gaddr) {
  asm volatile("global_load_async_to_lds_b128 %0, %1, off"
               :: "v"(lds_off_bytes), "v"(gaddr) : "memory");
}
__device__ __forceinline__ void wait_async_le4() {
  asm volatile("s_wait_asynccnt 0x4" ::: "memory");
}
__device__ __forceinline__ void wait_async_0() {
  asm volatile("s_wait_asynccnt 0x0" ::: "memory");
}

// ---------- global-direct fragment loaders (row-major, ld in elements) ----------
// A 16x32 bf16: lanes 0-15 -> K 0..7 (V0-3), 16..23 (V4-7); lanes 16-31 -> K 8..15, 24..31
__device__ __forceinline__ v16bf load_frag_a(const unsigned short* p, int row0, int k0, int ld) {
  int l = threadIdx.x & 31;
  int m = l & 15, kg = l >> 4;
  const unsigned short* q = p + (size_t)(row0 + m) * ld + k0 + 8 * kg;
  v4u lo = *(const v4u*)q;
  v4u hi = *(const v4u*)(q + 16);
  FragB16 f;
  f.u[0]=lo[0]; f.u[1]=lo[1]; f.u[2]=lo[2]; f.u[3]=lo[3];
  f.u[4]=hi[0]; f.u[5]=hi[1]; f.u[6]=hi[2]; f.u[7]=hi[3];
  return f.b;
}
// B 32x16 bf16 from Y[n][k] (n-major): lanes 0-15 -> K 0..7, 16..23; lanes 16-31 -> K 8..15, 24..31
__device__ __forceinline__ v16bf load_frag_b(const unsigned short* p, int n0, int k0, int ld) {
  int l = threadIdx.x & 31;
  int n = l & 15, kg = l >> 4;
  const unsigned short* q = p + (size_t)(n0 + n) * ld + k0 + 8 * kg;
  v4u lo = *(const v4u*)q;
  v4u hi = *(const v4u*)(q + 16);
  FragB16 f;
  f.u[0]=lo[0]; f.u[1]=lo[1]; f.u[2]=lo[2]; f.u[3]=lo[3];
  f.u[4]=hi[0]; f.u[5]=hi[1]; f.u[6]=hi[2]; f.u[7]=hi[3];
  return f.b;
}

// ---------- LDS fragment loaders (tile rows of ROWS ushorts) ----------
__device__ __forceinline__ v16bf lds_frag_a(const unsigned short* tile, int m0local) {
  int l = threadIdx.x & 31;
  int m = l & 15, kg = l >> 4;
  const unsigned short* p = tile + (m0local + m) * ROWS + 8 * kg;
  v4u lo = *(const v4u*)p;
  v4u hi = *(const v4u*)(p + 16);
  FragB16 f;
  f.u[0]=lo[0]; f.u[1]=lo[1]; f.u[2]=lo[2]; f.u[3]=lo[3];
  f.u[4]=hi[0]; f.u[5]=hi[1]; f.u[6]=hi[2]; f.u[7]=hi[3];
  return f.b;
}
__device__ __forceinline__ v16bf lds_frag_b(const unsigned short* tile, int n0local) {
  int l = threadIdx.x & 31;
  int n = l & 15, kg = l >> 4;
  const unsigned short* p = tile + (n0local + n) * ROWS + 8 * kg;
  v4u lo = *(const v4u*)p;
  v4u hi = *(const v4u*)(p + 16);
  FragB16 f;
  f.u[0]=lo[0]; f.u[1]=lo[1]; f.u[2]=lo[2]; f.u[3]=lo[3];
  f.u[4]=hi[0]; f.u[5]=hi[1]; f.u[6]=hi[2]; f.u[7]=hi[3];
  return f.b;
}

// C/D 16x16 f32: lane n = l&15; VGPR r holds row m = r + 8*(l>>4)
__device__ __forceinline__ void store_c_f32(float* p, v8f c, int m0, int n0, int ld) {
  int l = threadIdx.x & 31;
  int n = l & 15, mg = l >> 4;
  float* q = p + (size_t)(m0 + 8 * mg) * ld + n0 + n;
#pragma unroll
  for (int r = 0; r < 8; ++r) q[(size_t)r * ld] = c[r];
}
__device__ __forceinline__ void store_c_bf16(unsigned short* p, v8f c, int m0, int n0, int ld) {
  int l = threadIdx.x & 31;
  int n = l & 15, mg = l >> 4;
  unsigned short* q = p + (size_t)(m0 + 8 * mg) * ld + n0 + n;
#pragma unroll
  for (int r = 0; r < 8; ++r) q[(size_t)r * ld] = f32_to_bf16(c[r]);
}

#define WMMA_BF16(A, B, C) \
  __builtin_amdgcn_wmma_f32_16x16x32_bf16(false, (A), false, (B), (short)0, (C), false, false)

// stage one 128x32 A tile + 128x32 B tile into LDS (4 async b128 per thread)
__device__ __forceinline__ void stage_tiles(const unsigned short* A, const unsigned short* B,
                                            int m0, int n0, int kk, int lda, int ldb,
                                            unsigned lds_base_bytes) {
  int t = threadIdx.x;
#pragma unroll
  for (int j = 0; j < 2; ++j) {
    int c = t * 2 + j;                  // 0..511 : 128 rows x 4 chunks
    int row = c >> 2, col = c & 3;
    async_b128(lds_base_bytes + row * 80 + col * 16,
               A + (size_t)(m0 + row) * lda + kk + col * 8);
  }
#pragma unroll
  for (int j = 0; j < 2; ++j) {
    int c = t * 2 + j;
    int row = c >> 2, col = c & 3;
    async_b128(lds_base_bytes + TILE_S * 2 + row * 80 + col * 16,
               B + (size_t)(n0 + row) * ldb + kk + col * 8);
  }
}

// ---------- kernel 1: transpose+convert f32 [b][128][9216] -> bf16 [b][9216][128] ----------
__global__ __launch_bounds__(256) void efa_tcvt_kernel(const float* __restrict__ in,
                                                       unsigned short* __restrict__ out) {
  __shared__ float tile[32][33];
  int bid = blockIdx.x;
  int th = bid % (HW / 32); bid /= (HW / 32);
  int tc = bid % (CH / 32); bid /= (CH / 32);
  int b  = bid;
  int tx = threadIdx.x & 31, ty = threadIdx.x >> 5;
  const float* I = in + ((size_t)b * CH + tc * 32) * HW + (size_t)th * 32;
#pragma unroll
  for (int i = 0; i < 32; i += 8) tile[ty + i][tx] = I[(size_t)(ty + i) * HW + tx];
  __syncthreads();
  unsigned short* O = out + ((size_t)b * HW + th * 32) * CH + tc * 32;
#pragma unroll
  for (int i = 0; i < 32; i += 8) O[(size_t)(ty + i) * CH + tx] = f32_to_bf16(tile[tx][ty + i]);
}

// ---------- kernel 2: convert the 3 weight matrices to bf16 ----------
__global__ __launch_bounds__(256) void efa_cvtw_kernel(const float* __restrict__ wq,
                                                       const float* __restrict__ wk,
                                                       const float* __restrict__ wv,
                                                       unsigned short* __restrict__ out) {
  int i = blockIdx.x * 256 + threadIdx.x;
  const float* src = (i < 16384) ? wq : ((i < 32768) ? wk : wv);
  out[i] = f32_to_bf16(src[i & 16383]);
}

// ---------- kernel 3: projections q = Wq f, k = Wk e, v = Wv f (per batch) ----------
__global__ __launch_bounds__(256) void efa_proj_kernel(const unsigned short* __restrict__ Wb,
                                                       const unsigned short* __restrict__ fT,
                                                       const unsigned short* __restrict__ eT,
                                                       unsigned short* __restrict__ q,
                                                       unsigned short* __restrict__ k,
                                                       unsigned short* __restrict__ v) {
  int wave = threadIdx.x >> 5;
  int bid = blockIdx.x;
  int ng  = bid % 72; bid /= 72;
  int mt  = bid % 8;  bid /= 8;
  int b   = bid % 16; bid /= 16;
  int mat = bid;
  const unsigned short* W = Wb + (size_t)mat * CH * CH;
  const unsigned short* X = ((mat == 1) ? eT : fT) + (size_t)b * HW * CH;
  unsigned short* O = (mat == 0) ? q : ((mat == 1) ? k : v);
  int nt = ng * 8 + wave;
  v8f acc = {};
#pragma unroll
  for (int kk = 0; kk < CH; kk += 32) {
    v16bf a  = load_frag_a(W, mt * 16, kk, CH);
    v16bf bb = load_frag_b(X, nt * 16, kk, CH);
    acc = WMMA_BF16(a, bb, acc);
  }
  store_c_bf16(O + (size_t)b * CH * HW, acc, mt * 16, nt * 16, HW);
}

// ---------- kernel 4: transpose bf16 v [2048][9216] -> Vt [9216][2048] ----------
__global__ __launch_bounds__(256) void efa_tbf16_kernel(const unsigned short* __restrict__ in,
                                                        unsigned short* __restrict__ out) {
  __shared__ unsigned short tile[32][34];
  int bid = blockIdx.x;
  int tn = bid % (HW / 32); bid /= (HW / 32);
  int tm = bid;
  int tx = threadIdx.x & 31, ty = threadIdx.x >> 5;
  const unsigned short* I = in + ((size_t)tm * 32) * HW + (size_t)tn * 32;
#pragma unroll
  for (int i = 0; i < 32; i += 8) tile[ty + i][tx] = I[(size_t)(ty + i) * HW + tx];
  __syncthreads();
  unsigned short* O = out + ((size_t)tn * 32) * MTOT + tm * 32;
#pragma unroll
  for (int i = 0; i < 32; i += 8) O[(size_t)(ty + i) * MTOT + tx] = tile[tx][ty + i];
}

// ---------- staged 128x128 GEMM mainloop (async global->LDS double-buffered) ----------
__device__ __forceinline__ void gemm_mainloop(const unsigned short* __restrict__ A,
                                              const unsigned short* __restrict__ B,
                                              int m0, int n0, int lda, int ldb, int ktot,
                                              unsigned short* smem, v8f acc[8]) {
  unsigned base = (unsigned)(size_t)&smem[0];          // LDS byte offset (addr[31:0])
  int wave = threadIdx.x >> 5;
  int nstep = ktot / 32;
  stage_tiles(A, B, m0, n0, 0, lda, ldb, base);
  for (int t = 0; t < nstep; ++t) {
    int cur = t & 1;
    if (t + 1 < nstep) {
      stage_tiles(A, B, m0, n0, (t + 1) * 32, lda, ldb, base + (cur ^ 1) * (unsigned)(BUF_S * 2));
      wait_async_le4();                                // current buffer's copies are done
    } else {
      wait_async_0();
    }
    __syncthreads();                                   // all waves' tiles visible
    const unsigned short* As = smem + cur * BUF_S;
    const unsigned short* Bs = As + TILE_S;
    v16bf a = lds_frag_a(As, wave * 16);
#pragma unroll
    for (int j = 0; j < 8; ++j) {
      v16bf bb = lds_frag_b(Bs, 16 * j);
      acc[j] = WMMA_BF16(a, bb, acc[j]);
    }
    __syncthreads();                                   // done reading before re-staging
  }
}

// ---------- kernel 5: sim = Q * K^T   (2048 x 2048, K = 9216) ----------
__global__ __launch_bounds__(256) void efa_sim_kernel(const unsigned short* __restrict__ Q,
                                                      const unsigned short* __restrict__ Km,
                                                      float* __restrict__ sim) {
  __shared__ alignas(16) unsigned short smem[2 * BUF_S];
  int wave = threadIdx.x >> 5;
  int bn = blockIdx.x % (MTOT / 128);
  int bm = blockIdx.x / (MTOT / 128);
  int m0 = bm * 128, n0 = bn * 128;
  v8f acc[8] = {};
  gemm_mainloop(Q, Km, m0, n0, HW, HW, HW, smem, acc);
#pragma unroll
  for (int j = 0; j < 8; ++j) store_c_f32(sim, acc[j], m0 + wave * 16, n0 + 16 * j, MTOT);
}

// ---------- kernel 6: row softmax of sim, output bf16 ----------
__global__ __launch_bounds__(256) void efa_softmax_kernel(const float* __restrict__ sim,
                                                          unsigned short* __restrict__ simb) {
  __shared__ float red[256];
  int row = blockIdx.x;
  const float* s = sim + (size_t)row * MTOT;
  float vals[8];
  float mx = -3.4e38f;
#pragma unroll
  for (int j = 0; j < 8; ++j) { vals[j] = s[threadIdx.x + 256 * j]; mx = fmaxf(mx, vals[j]); }
  red[threadIdx.x] = mx; __syncthreads();
  for (int st = 128; st > 0; st >>= 1) {
    if (threadIdx.x < st) red[threadIdx.x] = fmaxf(red[threadIdx.x], red[threadIdx.x + st]);
    __syncthreads();
  }
  mx = red[0]; __syncthreads();
  float sum = 0.f;
#pragma unroll
  for (int j = 0; j < 8; ++j) { vals[j] = __expf(vals[j] - mx); sum += vals[j]; }
  red[threadIdx.x] = sum; __syncthreads();
  for (int st = 128; st > 0; st >>= 1) {
    if (threadIdx.x < st) red[threadIdx.x] += red[threadIdx.x + st];
    __syncthreads();
  }
  float inv = 1.0f / red[0];
#pragma unroll
  for (int j = 0; j < 8; ++j)
    simb[(size_t)row * MTOT + threadIdx.x + 256 * j] = f32_to_bf16(vals[j] * inv);
}

// ---------- kernel 7: out = simb * V + f_in   (2048 x 9216, K = 2048) ----------
__global__ __launch_bounds__(256) void efa_out_kernel(const unsigned short* __restrict__ S,
                                                      const unsigned short* __restrict__ Vt,
                                                      const float* __restrict__ f_in,
                                                      float* __restrict__ out) {
  __shared__ alignas(16) unsigned short smem[2 * BUF_S];
  int wave = threadIdx.x >> 5;
  int bn = blockIdx.x % (HW / 128);
  int bm = blockIdx.x / (HW / 128);
  int m0 = bm * 128, n0 = bn * 128;
  v8f acc[8] = {};
  gemm_mainloop(S, Vt, m0, n0, MTOT, MTOT, MTOT, smem, acc);
  int l = threadIdx.x & 31;
  int n = l & 15, mg = l >> 4;
#pragma unroll
  for (int j = 0; j < 8; ++j) {
#pragma unroll
    for (int r = 0; r < 8; ++r) {
      size_t idx = (size_t)(m0 + wave * 16 + 8 * mg + r) * HW + n0 + 16 * j + n;
      out[idx] = acc[j][r] + f_in[idx];
    }
  }
}

// ---------- launch ----------
extern "C" void kernel_launch(void* const* d_in, const int* in_sizes, int n_in,
                              void* d_out, int out_size, void* d_ws, size_t ws_size,
                              hipStream_t stream) {
  const float* f_in = (const float*)d_in[0];
  const float* e_in = (const float*)d_in[1];
  const float* Wq   = (const float*)d_in[2];
  const float* Wk   = (const float*)d_in[3];
  const float* Wv   = (const float*)d_in[4];
  float* out = (float*)d_out;
  char* ws = (char*)d_ws;

  const size_t SZ_T   = (size_t)BATCH * HW * CH * 2;
  const size_t SZ_QKV = (size_t)MTOT * HW * 2;
  const size_t OFS_FT  = 0;
  const size_t OFS_ET  = OFS_FT + SZ_T;
  const size_t OFS_WB  = OFS_ET + SZ_T;
  const size_t OFS_Q   = OFS_WB + 98304;
  const size_t OFS_K   = OFS_Q + SZ_QKV;
  const size_t OFS_V   = OFS_K + SZ_QKV;
  const size_t OFS_VT  = OFS_FT;                            // fT dead after proj
  const size_t OFS_SIM = OFS_ET;                            // eT dead after proj
  const size_t OFS_SIMB= OFS_ET + (size_t)MTOT * MTOT * 4;  // inside eT region

  unsigned short* fT   = (unsigned short*)(ws + OFS_FT);
  unsigned short* eT   = (unsigned short*)(ws + OFS_ET);
  unsigned short* Wb   = (unsigned short*)(ws + OFS_WB);
  unsigned short* q    = (unsigned short*)(ws + OFS_Q);
  unsigned short* k    = (unsigned short*)(ws + OFS_K);
  unsigned short* v    = (unsigned short*)(ws + OFS_V);
  unsigned short* Vt   = (unsigned short*)(ws + OFS_VT);
  float*          sim  = (float*)(ws + OFS_SIM);
  unsigned short* simb = (unsigned short*)(ws + OFS_SIMB);

  efa_tcvt_kernel<<<BATCH * (CH / 32) * (HW / 32), 256, 0, stream>>>(f_in, fT);
  efa_tcvt_kernel<<<BATCH * (CH / 32) * (HW / 32), 256, 0, stream>>>(e_in, eT);
  efa_cvtw_kernel<<<192, 256, 0, stream>>>(Wq, Wk, Wv, Wb);
  efa_proj_kernel<<<3 * 16 * 8 * 72, 256, 0, stream>>>(Wb, fT, eT, q, k, v);
  efa_tbf16_kernel<<<(MTOT / 32) * (HW / 32), 256, 0, stream>>>(v, Vt);
  efa_sim_kernel<<<(MTOT / 128) * (MTOT / 128), 256, 0, stream>>>(q, k, sim);
  efa_softmax_kernel<<<MTOT, 256, 0, stream>>>(sim, simb);
  efa_out_kernel<<<(MTOT / 128) * (HW / 128), 256, 0, stream>>>(simb, Vt, f_in, out);
}
